// VectorQuantizerEMAKeras_26800595927612
// MI455X (gfx1250) — matrix operand order repaired and checked by reading, not compiled, not run on hardware
//
#include <hip/hip_runtime.h>
#include <stdint.h>

// ---- problem constants -------------------------------------------------
#define NROWS 262144   // 64*64*64
#define DIM   128
#define KCB   1024
#define NCHUNK 64      // K / 16
#define MAINBLOCKS 2048 // NROWS / 128

typedef __attribute__((ext_vector_type(16))) __bf16 v16bf;
typedef __attribute__((ext_vector_type(8)))  float  v8f;

__device__ __forceinline__ v8f wmma_bf16(v16bf a, v16bf b, v8f c) {
  // D = A(16x32 bf16) x B(32x16 bf16) + C(16x16 f32)
  return __builtin_amdgcn_wmma_f32_16x16x32_bf16(false, a, false, b,
                                                 (short)0, c, false, false);
}

// async global->LDS copy of 16 bytes (CDNA5 GLOBAL_LOAD_ASYNC_TO_LDS_B128,
// tracked by ASYNCcnt). ldsaddr = 32-bit LDS byte address, gaddr = 64-bit VA.
__device__ __forceinline__ void async_copy_b128(unsigned ldsaddr, const void* gaddr) {
  asm volatile("global_load_async_to_lds_b128 %0, %1, off"
               :: "v"(ldsaddr), "v"(gaddr) : "memory");
}
__device__ __forceinline__ void wait_asynccnt0() {
  asm volatile("s_wait_asynccnt 0x0" ::: "memory");
}

// ---- stage 1: column norms of E ---------------------------------------
__global__ __launch_bounds__(256) void vq_prep_enorm(const float* __restrict__ E,
                                                     float* __restrict__ enorm) {
  int k = blockIdx.x * 256 + threadIdx.x;            // 0..1023
  float s = 0.f;
  #pragma unroll 8
  for (int d = 0; d < DIM; ++d) {
    float v = E[d * KCB + k];
    s = fmaf(v, v, s);
  }
  enorm[k] = s;
}

// ---- stage 2: swizzle E into WMMA B-fragment layout (bf16 hi + lo) ----
// B frag (32x16, 16-bit): lane l holds column n = base + (l&15),
// k = kk*32 + 8*(l>>4) + (j<8 ? j : j+8) for vector element j (ISA 7.12.2).
// Eswz[(chunk*32 + lane)*16 + j], chunk = c*4 + kk, 256 chunks total.
__global__ __launch_bounds__(256) void vq_prep_swz(const float* __restrict__ E,
                                                   unsigned short* __restrict__ EH,
                                                   unsigned short* __restrict__ EL) {
  int t = blockIdx.x * 256 + threadIdx.x;            // 0..8191
  int chunk = t >> 5, lane = t & 31;
  int c = chunk >> 2, kk = chunk & 3;
  int n  = c * 16 + (lane & 15);
  int kb = kk * 32 + (lane >> 4) * 8;
  #pragma unroll
  for (int j = 0; j < 16; ++j) {
    float v   = E[(kb + (j < 8 ? j : j + 8)) * KCB + n];
    __bf16 h  = (__bf16)v;
    __bf16 lo = (__bf16)(v - (float)h);
    EH[t * 16 + j] = __builtin_bit_cast(unsigned short, h);
    EL[t * 16 + j] = __builtin_bit_cast(unsigned short, lo);
  }
}

// ---- stage 3: ET[k][d] = E[d][k]  (contiguous codebook rows for gather)
__global__ __launch_bounds__(256) void vq_prep_et(const float* __restrict__ E,
                                                  float* __restrict__ ET) {
  int t = blockIdx.x * 256 + threadIdx.x;            // 0..131071
  int k = t >> 7, d = t & 127;
  ET[k * DIM + d] = E[d * KCB + k];
}

// ---- stage 4: fused distance-GEMM + argmin + gather + loss ------------
__global__ __launch_bounds__(256) void vq_main(const float* __restrict__ X,
                                               const float* __restrict__ ET,
                                               const unsigned short* __restrict__ EH,
                                               const unsigned short* __restrict__ EL,
                                               const float* __restrict__ enorm,
                                               float* __restrict__ qout,
                                               float* __restrict__ idxout,
                                               float* __restrict__ partials) {
  __shared__ __align__(16) unsigned short sB[2][2][2048]; // [buf][hi/lo][16-col chunk]
  __shared__ float sEn[KCB];
  __shared__ float sRed[8];

  const int tid  = threadIdx.x;
  const int wave = tid >> 5;
  const int lane = tid & 31;
  const int hl   = lane >> 4;      // lane half (ISA A/B layout)
  const int lm   = lane & 15;
  const int rowbase = blockIdx.x * 128 + wave * 16;

  // codebook norms -> LDS (covered by first barrier below)
  #pragma unroll
  for (int i = 0; i < 4; ++i) sEn[tid + i * 256] = enorm[tid + i * 256];

  // ---- build A fragments in registers: row m = lm, split hi/lo bf16 ----
  // 16-bit A 16x32: element j -> k = kk*32 + hl*8 + (j<8 ? j : j+8)
  const float* xr = X + (size_t)(rowbase + lm) * DIM;
  v16bf aH[4], aL[4];
  #pragma unroll
  for (int kk = 0; kk < 4; ++kk) {
    const float* p = xr + kk * 32 + hl * 8;
    float v[16];
    float4 f0 = *(const float4*)(p);
    float4 f1 = *(const float4*)(p + 4);
    float4 f2 = *(const float4*)(p + 16);
    float4 f3 = *(const float4*)(p + 20);
    v[0]=f0.x; v[1]=f0.y; v[2]=f0.z; v[3]=f0.w;
    v[4]=f1.x; v[5]=f1.y; v[6]=f1.z; v[7]=f1.w;
    v[8]=f2.x; v[9]=f2.y; v[10]=f2.z; v[11]=f2.w;
    v[12]=f3.x; v[13]=f3.y; v[14]=f3.z; v[15]=f3.w;
    #pragma unroll
    for (int j = 0; j < 16; ++j) {
      __bf16 h = (__bf16)v[j];
      aH[kk][j] = h;
      aL[kk][j] = (__bf16)(v[j] - (float)h);
    }
  }

  // per-thread LDS byte addresses of the two staging buffers
  const unsigned ldsH0 = (unsigned)(uintptr_t)&sB[0][0][tid * 8];
  const unsigned ldsL0 = (unsigned)(uintptr_t)&sB[0][1][tid * 8];
  const unsigned ldsH1 = (unsigned)(uintptr_t)&sB[1][0][tid * 8];
  const unsigned ldsL1 = (unsigned)(uintptr_t)&sB[1][1][tid * 8];

  // ---- prologue: async-stage chunk 0 into LDS buffer 0 -----------------
  async_copy_b128(ldsH0, EH + tid * 8);
  async_copy_b128(ldsL0, EL + tid * 8);
  wait_asynccnt0();
  __syncthreads();

  float best[8];
  int   bidx[8];
  #pragma unroll
  for (int v = 0; v < 8; ++v) { best[v] = 3.4e38f; bidx[v] = 0; }

  // ---- main loop over 64 column chunks of 16 ---------------------------
  for (int c = 0; c < NCHUNK; ++c) {
    const int buf = c & 1;
    if (c < NCHUNK - 1) {              // async prefetch of next chunk -> other buf
      async_copy_b128(buf ? ldsH0 : ldsH1, EH + (c + 1) * 2048 + tid * 8);
      async_copy_b128(buf ? ldsL0 : ldsL1, EL + (c + 1) * 2048 + tid * 8);
    }
    v8f acc0 = {0,0,0,0,0,0,0,0};      // xh*eh chain
    v8f acc1 = {0,0,0,0,0,0,0,0};      // xl*eh + xh*el chain
    #pragma unroll
    for (int kk = 0; kk < 4; ++kk) {
      v16bf bH = *(const v16bf*)&sB[buf][0][(kk * 32 + lane) * 16];
      v16bf bL = *(const v16bf*)&sB[buf][1][(kk * 32 + lane) * 16];
      acc0 = wmma_bf16(aH[kk], bH, acc0);
      acc1 = wmma_bf16(aL[kk], bH, acc1);
      acc1 = wmma_bf16(aH[kk], bL, acc1);
    }
    // score = ||e||^2 - 2*dot ; D layout: elem v -> row (v + 8*hl), col lm
    const int n  = c * 16 + lm;
    const float en = sEn[n];
    #pragma unroll
    for (int v = 0; v < 8; ++v) {
      float s = fmaf(-2.0f, acc0[v] + acc1[v], en);
      if (s < best[v]) { best[v] = s; bidx[v] = n; }
    }
    wait_asynccnt0();                  // staging of next chunk complete
    __syncthreads();
  }

  // ---- argmin reduction across the 16-lane column groups ---------------
  #pragma unroll
  for (int off = 1; off <= 8; off <<= 1) {
    #pragma unroll
    for (int v = 0; v < 8; ++v) {
      float os = __shfl_xor(best[v], off, 32);
      int   oi = __shfl_xor(bidx[v], off, 32);
      if (os < best[v] || (os == best[v] && oi < bidx[v])) {
        best[v] = os; bidx[v] = oi;
      }
    }
  }

  // ---- epilogue: gather quantized rows, indices, commitment loss -------
  float lossAcc = 0.f;
  int myBi = 0;
  #pragma unroll
  for (int r = 0; r < 16; ++r) {
    int bi = __shfl(bidx[r & 7], (r < 8) ? 0 : 16, 32);  // uniform in wave
    if (lane == r) myBi = bi;
    const size_t row = (size_t)(rowbase + r);
    float4 qv = *(const float4*)(ET + bi * DIM + lane * 4);
    float4 xv = *(const float4*)(X + row * DIM + lane * 4);
    float dx = qv.x - xv.x, dy = qv.y - xv.y;
    float dz = qv.z - xv.z, dw = qv.w - xv.w;
    lossAcc += dx * dx + dy * dy + dz * dz + dw * dw;
    *(float4*)(qout + row * DIM + lane * 4) = qv;
  }
  if (lane < 16) idxout[rowbase + lane] = (float)myBi;

  #pragma unroll
  for (int off = 16; off; off >>= 1) lossAcc += __shfl_xor(lossAcc, off, 32);
  if (lane == 0) sRed[wave] = lossAcc;
  __syncthreads();
  if (tid == 0) {
    float s = 0.f;
    #pragma unroll
    for (int w = 0; w < 8; ++w) s += sRed[w];
    partials[blockIdx.x] = s;   // fixed order -> deterministic
  }
}

// ---- stage 5: deterministic loss reduction ----------------------------
__global__ __launch_bounds__(256) void vq_final(const float* __restrict__ partials,
                                                float* __restrict__ lossp) {
  __shared__ float sm[256];
  int t = threadIdx.x;
  float s = 0.f;
  for (int i = t; i < MAINBLOCKS; i += 256) s += partials[i];
  sm[t] = s;
  __syncthreads();
  for (int off = 128; off; off >>= 1) {
    if (t < off) sm[t] += sm[t + off];
    __syncthreads();
  }
  if (t == 0) lossp[0] = 0.25f * sm[0] / 33554432.0f;  // mean over N*D
}

// ---- host launcher ----------------------------------------------------
extern "C" void kernel_launch(void* const* d_in, const int* in_sizes, int n_in,
                              void* d_out, int out_size, void* d_ws, size_t ws_size,
                              hipStream_t stream) {
  (void)in_sizes; (void)n_in; (void)out_size; (void)ws_size;
  const float* X = (const float*)d_in[0];          // [262144,128] f32
  const float* E = (const float*)d_in[1];          // [128,1024] f32

  float* out   = (float*)d_out;
  float* qout  = out;                               // [N*D]
  float* lossp = out + (size_t)NROWS * DIM;         // [1]
  float* idxp  = lossp + 1;                         // [N] (as float)

  char* ws = (char*)d_ws;
  unsigned short* EH = (unsigned short*)ws;                       // 256 KB
  unsigned short* EL = (unsigned short*)(ws + 262144);            // 256 KB
  float* ET          = (float*)(ws + 524288);                     // 512 KB
  float* enorm       = (float*)(ws + 524288 + 524288);            // 4 KB
  float* partials    = (float*)(ws + 524288 + 524288 + 4096);     // 8 KB

  vq_prep_enorm<<<4,   256, 0, stream>>>(E, enorm);
  vq_prep_swz  <<<32,  256, 0, stream>>>(E, EH, EL);
  vq_prep_et   <<<512, 256, 0, stream>>>(E, ET);
  vq_main      <<<MAINBLOCKS, 256, 0, stream>>>(X, ET, EH, EL, enorm,
                                                qout, idxp, partials);
  vq_final     <<<1,   256, 0, stream>>>(partials, lossp);
}
